// TinyDualStream_7086696038577
// MI455X (gfx1250) — matrix operand order repaired and checked by reading, not compiled, not run on hardware
//
#include <hip/hip_runtime.h>
#include <hip/hip_bf16.h>
#include <math.h>

// ---------------------------------------------------------------------------
// CDNA5 (gfx1250) implementation of the TinyDualStream point-cloud network.
// All GEMM-shaped work runs on v_wmma_f32_16x16x32_f16 (wave32 WMMA).
// Weights are pre-packed into WMMA B-fragment layout (f16, zero padded) by a
// prep kernel so the hot loops issue only unconditional b128 loads.
// Workspace layout (~85.2 MB; everything stays resident in the 192MB L2):
//   [0,   4MB)       : knn indices, int32 [256 clouds][256 pts][16]
//   [4,  20MB)       : local features, f16 [65536 rows][128]
//   [20, 84MB)       : h activations, f32 [65536 rows][256]
//   [84MB, +0.6MB)   : per-(b,t) 576-dim features, f32 [256][576]
//   [85MB, +160KB)   : packed weight fragments (ew1, ew2, p1w, p2w)
// ---------------------------------------------------------------------------

typedef __attribute__((ext_vector_type(16))) _Float16 v16h;
typedef __attribute__((ext_vector_type(8)))  _Float16 v8h;
typedef __attribute__((ext_vector_type(4)))  _Float16 h4;
typedef __attribute__((ext_vector_type(8)))  float    v8f;

// Branchless tanh-form GELU: 0.5x(1+tanh(sqrt(2/pi)(x+0.044715x^3))),
// tanh(u) = 1 - 2/(exp(2u)+1). Odd in u -> no abs/copysign; exp/rcp saturate
// correctly at +-inf. ~10 VALU ops, zero branches.
__device__ __forceinline__ float gelu_f(float x) {
  const float u2 = x * fmaf(0.07135481283f, x * x, 1.5957691216f); // 2u
  const float e  = __expf(u2);
  const float t  = fmaf(-2.0f, __builtin_amdgcn_rcpf(e + 1.0f), 1.0f);
  return 0.5f * x * (1.0f + t);
}

__device__ __forceinline__ v8f wmma_f16(v16h a, v16h b, v8f c) {
  return __builtin_amdgcn_wmma_f32_16x16x32_f16(
      false, a, false, b, (short)0, c, false, false);
}

// A-fragment K index for f16 16x32 A: slot s (0..15) of lane `lane`.
__device__ __forceinline__ int a_k(int lane, int s) {
  return ((s >> 3) << 4) + ((lane >> 4) << 3) + (s & 7);
}
// B-fragment K index for f16 32x16 B.
__device__ __forceinline__ int b_k(int lane, int s) {
  return ((s >> 3) << 4) + ((lane >> 4) << 3) + (s & 7);
}
__device__ __forceinline__ v16h cat8(v8h lo, v8h hi) {
  return __builtin_shufflevector(lo, hi, 0, 1, 2, 3, 4, 5, 6, 7,
                                 8, 9, 10, 11, 12, 13, 14, 15);
}

// ---------------------------------------------------------------------------
// Kernel 0: pack a weight matrix into WMMA B-fragment layout (f16, padded).
// GEMM view: B[k][n], k < K (zero beyond), n < N. If `transposed`, the source
// is stored [n][k] (out-major, e.g. ew1/ew2), else [k][n] (e.g. p1w/p2w).
// frag linear index = ((kc*(N/16) + t)*32 + lane)*16 + s.
// ---------------------------------------------------------------------------
__global__ void pack_wfrag_kernel(const float* __restrict__ w,
                                  _Float16* __restrict__ frag,
                                  int K, int N, int KC, int transposed) {
  const int tid = blockIdx.x * 256 + threadIdx.x;
  const int total = KC * (N >> 4) * 512;
  if (tid >= total) return;
  const int s = tid & 15;
  const int lane = (tid >> 4) & 31;
  const int t = (tid >> 9) % (N >> 4);
  const int kc = tid / ((N >> 4) << 9);
  const int k = kc * 32 + b_k(lane, s);
  const int n = t * 16 + (lane & 15);
  const int kk = (k < K) ? k : 0;                       // clamp, no branch
  const float v = transposed ? w[n * K + kk] : w[kk * N + n];
  frag[tid] = (_Float16)((k < K) ? v : 0.f);
}

// ---------------------------------------------------------------------------
// Kernel 1: kNN (k=16) over xyz per cloud. One block = one cloud of 256 pts.
// ---------------------------------------------------------------------------
__global__ void knn_kernel(const float* __restrict__ x_pt, int* __restrict__ idx_out) {
  const int cloud = blockIdx.x;
  const int p = threadIdx.x;
  __shared__ float sx[256][3];
  const float* xb = x_pt + (size_t)cloud * 256 * 10;
  sx[p][0] = xb[p * 10 + 0];
  sx[p][1] = xb[p * 10 + 1];
  sx[p][2] = xb[p * 10 + 2];
  __syncthreads();
  const float px = sx[p][0], py = sx[p][1], pz = sx[p][2];
  float bd[16];
  int bi[16];
#pragma unroll
  for (int j = 0; j < 16; ++j) { bd[j] = 3.4e38f; bi[j] = 0; }
  for (int q = 0; q < 256; ++q) {
    const float dx = sx[q][0] - px, dy = sx[q][1] - py, dz = sx[q][2] - pz;
    const float d = dx * dx + dy * dy + dz * dz;
    if (d < bd[15]) {
      int j = 15;
      while (j > 0 && bd[j - 1] > d) { bd[j] = bd[j - 1]; bi[j] = bi[j - 1]; --j; }
      bd[j] = d; bi[j] = q;
    }
  }
  int* o = idx_out + ((size_t)cloud * 256 + p) * 16;
#pragma unroll
  for (int j = 0; j < 16; ++j) o[j] = bi[j];
}

// ---------------------------------------------------------------------------
// Kernel 2: edge conv. One block = one cloud (8 waves). Each wave owns a
// point: its 16 edges are the 16 rows of a WMMA M-tile. 20->64->128 with
// BN+GELU, then max over the M dimension (neighbors).
// The fragment-table pointers are laundered through an empty volatile asm
// inside the point loop so LICM cannot hoist ~160 VGPRs of B fragments (which
// previously spilled to scratch); the per-iteration re-loads hit L0.
// ---------------------------------------------------------------------------
__global__ void edgeconv_kernel(
    const float* __restrict__ x_pt, const int* __restrict__ idx_in,
    const _Float16* __restrict__ ew1f, const _Float16* __restrict__ ew2f,
    const float* __restrict__ eg1, const float* __restrict__ eb1,
    const float* __restrict__ em1, const float* __restrict__ ev1,
    const float* __restrict__ eg2, const float* __restrict__ eb2,
    const float* __restrict__ em2, const float* __restrict__ ev2,
    _Float16* __restrict__ local_out) {
  const int cloud = blockIdx.x;
  __shared__ float sx[256][10];          // 10 KB
  __shared__ _Float16 a1[8][16][64];     // 16 KB per-wave layer1 activations
  __shared__ float sc1[64], sh1[64], sc2[128], sh2[128];
  const int tid = threadIdx.x;
  __builtin_prefetch(ew2f, 0, 0);        // global_prefetch_b8 into caches
  const float* xb = x_pt + (size_t)cloud * 2560;
#pragma unroll
  for (int j = 0; j < 10; ++j) sx[tid][j] = xb[tid * 10 + j];
  if (tid < 64) {
    const float sc = rsqrtf(ev1[tid] + 1e-5f) * eg1[tid];
    sc1[tid] = sc;
    sh1[tid] = eb1[tid] - em1[tid] * sc;
  }
  if (tid < 128) {
    const float sc = rsqrtf(ev2[tid] + 1e-5f) * eg2[tid];
    sc2[tid] = sc;
    sh2[tid] = eb2[tid] - em2[tid] * sc;
  }
  __syncthreads();

  const int wave = tid >> 5, lane = tid & 31;
  const int m = lane & 15;
  const int khalf = lane >> 4;
  const v8f vzero = {0.f, 0.f, 0.f, 0.f, 0.f, 0.f, 0.f, 0.f};

  for (int p = wave; p < 256; p += 8) {
    // Opaque copies of the fragment-table pointers: redefined every iteration
    // by the volatile asm, so dependent loads cannot be hoisted out of the
    // loop (they would spill). "s" keeps them in SGPRs (saddr addressing).
    const _Float16* ew1p = ew1f;
    const _Float16* ew2p = ew2f;
    asm volatile("" : "+s"(ew1p), "+s"(ew2p));

    const int nb = idx_in[((size_t)cloud * 256 + p) * 16 + m];
    // --- A fragment: edge features (branchless: clamped loads + selects) ---
    v16h A;
#pragma unroll
    for (int s = 0; s < 16; ++s) {
      const int k = a_k(lane, s);
      const int km = k - 10;
      const int kcl = (k < 10) ? k : ((km < 10) ? km : 0);
      const float xi = sx[p][kcl];
      const float xj = sx[nb][kcl];
      float v = (k < 10) ? xi : (xj - xi);
      v = (k < 20) ? v : 0.f;
      A[s] = (_Float16)v;
    }
    // --- Layer 1: N=64 (4 tiles) ------------------------------------------
#pragma unroll
    for (int t = 0; t < 4; ++t) {
      const int oc = t * 16 + m;
      const v16h B = *(const v16h*)&ew1p[(size_t)(t * 32 + lane) * 16];
      v8f C = vzero;
      C = wmma_f16(A, B, C);
      const float sc = sc1[oc], sh = sh1[oc];
#pragma unroll
      for (int g = 0; g < 8; ++g)
        a1[wave][g + (khalf << 3)][oc] = (_Float16)gelu_f(C[g] * sc + sh);
    }
    // --- Layer 2: K=64 (2 chunks), N=128 (8 tiles); max over M ------------
#pragma unroll
    for (int t = 0; t < 8; ++t) {
      const int oc = t * 16 + m;
      v8f C = vzero;
#pragma unroll
      for (int kc = 0; kc < 2; ++kc) {
        const v8h lo = *(const v8h*)&a1[wave][m][kc * 32 + khalf * 8];
        const v8h hi = *(const v8h*)&a1[wave][m][kc * 32 + 16 + khalf * 8];
        const v16h B2 = *(const v16h*)&ew2p[(size_t)((kc * 8 + t) * 32 + lane) * 16];
        C = wmma_f16(cat8(lo, hi), B2, C);
      }
      const float sc = sc2[oc], sh = sh2[oc];
      float mx = -3.4e38f;
#pragma unroll
      for (int g = 0; g < 8; ++g) mx = fmaxf(mx, gelu_f(C[g] * sc + sh));
      mx = fmaxf(mx, __shfl_xor(mx, 16, 32));
      if (lane < 16)
        local_out[((size_t)cloud * 256 + p) * 128 + oc] = (_Float16)mx;
    }
  }
}

// ---------------------------------------------------------------------------
// Kernel 3: per-point MLP. 138->128 (LN,GELU) -> 256 (LN,GELU).
// Block = 64 threads (2 waves); each wave owns one 16-row tile.
// ---------------------------------------------------------------------------
__global__ void pointmlp_kernel(
    const float* __restrict__ x_pt, const _Float16* __restrict__ local_in,
    const _Float16* __restrict__ p1wf, const float* __restrict__ p1b,
    const float* __restrict__ ln1g, const float* __restrict__ ln1b,
    const _Float16* __restrict__ p2wf, const float* __restrict__ p2b,
    const float* __restrict__ ln2g, const float* __restrict__ ln2b,
    float* __restrict__ h_out) {
  __shared__ float    rows1[2][16][128];   // 16 KB
  __shared__ _Float16 act1 [2][16][128];   //  8 KB
  __shared__ float    rows2[2][16][256];   // 32 KB
  const int tid = threadIdx.x, wave = tid >> 5, lane = tid & 31;
  const int tile = blockIdx.x * 2 + wave;
  const int m = lane & 15, khalf = lane >> 4;
  const int rowbase = tile * 16;
  const v8f vzero = {0.f, 0.f, 0.f, 0.f, 0.f, 0.f, 0.f, 0.f};
  __builtin_prefetch(p2wf, 0, 0);

  // ---- Layer 1: K = 138 padded to 160 (5 chunks), N = 128 (8 tiles) -------
  v8f C[8];
#pragma unroll
  for (int t = 0; t < 8; ++t) C[t] = vzero;
  const int row = rowbase + m;
  const _Float16* lrow = local_in + (size_t)row * 128;
  for (int kc = 0; kc < 5; ++kc) {
    v16h A;
    if (kc < 4) {                       // uniform branch: k always < 128
      const v8h lo = *(const v8h*)&lrow[kc * 32 + khalf * 8];
      const v8h hi = *(const v8h*)&lrow[kc * 32 + 16 + khalf * 8];
      A = cat8(lo, hi);
    } else {                            // tail chunk: 10 xyz-feature columns
#pragma unroll
      for (int s = 0; s < 16; ++s) {
        const int k = a_k(lane, s);     // 0..31 within chunk (global 128+k)
        const int kk = (k < 10) ? k : 0;
        const float v = x_pt[(size_t)row * 10 + kk];
        A[s] = (_Float16)((k < 10) ? v : 0.f);
      }
    }
#pragma unroll
    for (int t = 0; t < 8; ++t) {
      const v16h B = *(const v16h*)&p1wf[(size_t)((kc * 8 + t) * 32 + lane) * 16];
      C[t] = wmma_f16(A, B, C[t]);
    }
  }
#pragma unroll
  for (int t = 0; t < 8; ++t) {
    const int col = t * 16 + m;
#pragma unroll
    for (int g = 0; g < 8; ++g)
      rows1[wave][g + (khalf << 3)][col] = C[t][g] + p1b[col];
  }
  // ---- LN(128) + GELU -> act1 (f16, packed h4 stores) ---------------------
  {
    const int rr = lane >> 1, hh = lane & 1;
    const float4* r4 = (const float4*)&rows1[wave][rr][hh * 64];
    float s0 = 0.f;
#pragma unroll
    for (int j = 0; j < 16; ++j) {
      const float4 v = r4[j];
      s0 += (v.x + v.y) + (v.z + v.w);
    }
    s0 += __shfl_xor(s0, 1, 32);
    const float mu = s0 * (1.f / 128.f);
    float v0 = 0.f;
#pragma unroll
    for (int j = 0; j < 16; ++j) {
      const float4 v = r4[j];
      const float a = v.x - mu, b = v.y - mu, cc = v.z - mu, dd = v.w - mu;
      v0 += (a * a + b * b) + (cc * cc + dd * dd);
    }
    v0 += __shfl_xor(v0, 1, 32);
    const float rs = rsqrtf(v0 * (1.f / 128.f) + 1e-5f);
#pragma unroll
    for (int j = 0; j < 16; ++j) {
      const int col = hh * 64 + j * 4;
      const float4 v = r4[j];
      const float4 gv = *(const float4*)&ln1g[col];
      const float4 bv = *(const float4*)&ln1b[col];
      h4 o;
      o[0] = (_Float16)gelu_f((v.x - mu) * rs * gv.x + bv.x);
      o[1] = (_Float16)gelu_f((v.y - mu) * rs * gv.y + bv.y);
      o[2] = (_Float16)gelu_f((v.z - mu) * rs * gv.z + bv.z);
      o[3] = (_Float16)gelu_f((v.w - mu) * rs * gv.w + bv.w);
      *(h4*)&act1[wave][rr][col] = o;
    }
  }
  // ---- Layer 2: K = 128 (4 chunks), N = 256 (two halves of 8 tiles) -------
  for (int nh = 0; nh < 2; ++nh) {
    v8f C2[8];
#pragma unroll
    for (int t = 0; t < 8; ++t) C2[t] = vzero;
    for (int kc = 0; kc < 4; ++kc) {
      const v8h lo = *(const v8h*)&act1[wave][m][kc * 32 + khalf * 8];
      const v8h hi = *(const v8h*)&act1[wave][m][kc * 32 + 16 + khalf * 8];
      const v16h A2 = cat8(lo, hi);
#pragma unroll
      for (int t = 0; t < 8; ++t) {
        const int tt = nh * 8 + t;
        const v16h B2 = *(const v16h*)&p2wf[(size_t)((kc * 16 + tt) * 32 + lane) * 16];
        C2[t] = wmma_f16(A2, B2, C2[t]);
      }
    }
#pragma unroll
    for (int t = 0; t < 8; ++t) {
      const int col = (nh * 8 + t) * 16 + m;
#pragma unroll
      for (int g = 0; g < 8; ++g)
        rows2[wave][g + (khalf << 3)][col] = C2[t][g] + p2b[col];
    }
  }
  // ---- LN(256) + GELU -> h_out (f32, float4 stores) -----------------------
  {
    const int rr = lane >> 1, hh = lane & 1;
    const float4* r4 = (const float4*)&rows2[wave][rr][hh * 128];
    float s0 = 0.f;
#pragma unroll
    for (int j = 0; j < 32; ++j) {
      const float4 v = r4[j];
      s0 += (v.x + v.y) + (v.z + v.w);
    }
    s0 += __shfl_xor(s0, 1, 32);
    const float mu = s0 * (1.f / 256.f);
    float v0 = 0.f;
#pragma unroll
    for (int j = 0; j < 32; ++j) {
      const float4 v = r4[j];
      const float a = v.x - mu, b = v.y - mu, cc = v.z - mu, dd = v.w - mu;
      v0 += (a * a + b * b) + (cc * cc + dd * dd);
    }
    v0 += __shfl_xor(v0, 1, 32);
    const float rs = rsqrtf(v0 * (1.f / 256.f) + 1e-5f);
    float4* o4 = (float4*)(h_out + (size_t)(rowbase + rr) * 256);
#pragma unroll
    for (int j = 0; j < 32; ++j) {
      const int col = hh * 128 + j * 4;
      const float4 v = r4[j];
      const float4 gv = *(const float4*)&ln2g[col];
      const float4 bv = *(const float4*)&ln2b[col];
      float4 o;
      o.x = gelu_f((v.x - mu) * rs * gv.x + bv.x);
      o.y = gelu_f((v.y - mu) * rs * gv.y + bv.y);
      o.z = gelu_f((v.z - mu) * rs * gv.z + bv.z);
      o.w = gelu_f((v.w - mu) * rs * gv.w + bv.w);
      o4[col >> 2] = o;
    }
  }
}

// ---------------------------------------------------------------------------
// Kernel 4: max+mean pool over P (channel per thread) + frame MLP 4->32->64.
// ---------------------------------------------------------------------------
__global__ void pool_fr_kernel(
    const float* __restrict__ h, const float* __restrict__ x_fr,
    const float* __restrict__ f1w, const float* __restrict__ f1b,
    const float* __restrict__ f2w, const float* __restrict__ f2b,
    float* __restrict__ per) {
  const int bt = blockIdx.x, c = threadIdx.x;
  const float* hb = h + (size_t)bt * 256 * 256;
  float mx = -3.4e38f, sm = 0.f;
  for (int p = 0; p < 256; ++p) {
    const float v = hb[(size_t)p * 256 + c];
    mx = fmaxf(mx, v);
    sm += v;
  }
  per[(size_t)bt * 576 + c] = mx;
  per[(size_t)bt * 576 + 256 + c] = sm * (1.f / 256.f);
  __shared__ float hid[32];
  if (c < 32) {
    float a = f1b[c];
#pragma unroll
    for (int i = 0; i < 4; ++i) a += x_fr[bt * 4 + i] * f1w[i * 32 + c];
    hid[c] = gelu_f(a);
  }
  __syncthreads();
  if (c < 64) {
    float a = f2b[c];
    for (int j = 0; j < 32; ++j) a += hid[j] * f2w[j * 64 + c];
    per[(size_t)bt * 576 + 512 + c] = a;
  }
}

// ---------------------------------------------------------------------------
// Kernel 5: 1x1 proj + 4 conv1d(k=3) blocks + pool + head. One block per b.
// hs is [channel][t] padded to 36 so each row is 9 aligned float4 LDS loads;
// halo (t=0) and tail (t=33..35) are zero. Heavy loops pinned rolled to keep
// static code size in check (dynamic work unchanged).
// ---------------------------------------------------------------------------
__device__ __forceinline__ void conv_block(
    float (*hs)[36], const float* __restrict__ w, const float* __restrict__ bias,
    const float* __restrict__ g, const float* __restrict__ be,
    const float* __restrict__ m, const float* __restrict__ v,
    int c, bool residual) {
  float acc[32];
#pragma unroll
  for (int t = 0; t < 32; ++t) acc[t] = bias[c];
  const float4* w4 = (const float4*)(w + (size_t)c * 768);   // 16B aligned
#pragma unroll 1
  for (int i4 = 0; i4 < 64; ++i4) {          // 4 input channels per iteration
    const float4 wa = w4[i4 * 3 + 0];
    const float4 wb = w4[i4 * 3 + 1];
    const float4 wc = w4[i4 * 3 + 2];
    const float wgt[12] = {wa.x, wa.y, wa.z, wa.w, wb.x, wb.y, wb.z, wb.w,
                           wc.x, wc.y, wc.z, wc.w};
#pragma unroll
    for (int ii = 0; ii < 4; ++ii) {
      const int i = i4 * 4 + ii;
      float r[36];
      const float4* h4p = (const float4*)hs[i];
#pragma unroll
      for (int q = 0; q < 9; ++q) {
        const float4 x = h4p[q];
        r[q * 4 + 0] = x.x; r[q * 4 + 1] = x.y;
        r[q * 4 + 2] = x.z; r[q * 4 + 3] = x.w;
      }
      const float w0 = wgt[ii * 3 + 0];
      const float w1 = wgt[ii * 3 + 1];
      const float w2 = wgt[ii * 3 + 2];
#pragma unroll
      for (int t = 0; t < 32; ++t)
        acc[t] += w0 * r[t] + w1 * r[t + 1] + w2 * r[t + 2];
    }
  }
  const float sc = rsqrtf(v[c] + 1e-5f) * g[c];
  const float sh = be[c] - m[c] * sc;
  __syncthreads();
#pragma unroll
  for (int t = 0; t < 32; ++t) {
    const float y = gelu_f(acc[t] * sc + sh);
    hs[c][t + 1] = residual ? hs[c][t + 1] + y : y;
  }
  __syncthreads();
}

__global__ void conv_head_kernel(
    const float* __restrict__ per,
    const float* __restrict__ projw, const float* __restrict__ projb,
    const float* __restrict__ c1w, const float* __restrict__ c1b,
    const float* __restrict__ g1, const float* __restrict__ be1,
    const float* __restrict__ m1, const float* __restrict__ v1,
    const float* __restrict__ c2w, const float* __restrict__ c2b,
    const float* __restrict__ g2, const float* __restrict__ be2,
    const float* __restrict__ m2, const float* __restrict__ v2,
    const float* __restrict__ c3w, const float* __restrict__ c3b,
    const float* __restrict__ g3, const float* __restrict__ be3,
    const float* __restrict__ m3, const float* __restrict__ v3,
    const float* __restrict__ c4w, const float* __restrict__ c4b,
    const float* __restrict__ g4, const float* __restrict__ be4,
    const float* __restrict__ m4, const float* __restrict__ v4,
    const float* __restrict__ h1w, const float* __restrict__ h1b,
    const float* __restrict__ h2w, const float* __restrict__ h2b,
    float* __restrict__ out) {
  const int b = blockIdx.x, c = threadIdx.x;
  __shared__ float hs[256][36];
  __shared__ float stage[576];
  __shared__ float pooled[256];
  __shared__ float hid[128];
  hs[c][0] = 0.f;
  hs[c][33] = 0.f;
  hs[c][34] = 0.f;
  hs[c][35] = 0.f;
  // 1x1 projection 576 -> 256 (bias only), float4 dot products.
  const float4* pw4 = (const float4*)(projw + (size_t)c * 576);
  for (int t = 0; t < 32; ++t) {
    for (int e = c; e < 576; e += 256)
      stage[e] = per[((size_t)b * 32 + t) * 576 + e];
    __syncthreads();
    const float4* st4 = (const float4*)stage;
    float a = projb[c];
#pragma unroll 1
    for (int e = 0; e < 144; ++e) {
      const float4 sv = st4[e];
      const float4 wv = pw4[e];
      a += (sv.x * wv.x + sv.y * wv.y) + (sv.z * wv.z + sv.w * wv.w);
    }
    hs[c][t + 1] = a;
    __syncthreads();
  }
  conv_block(hs, c1w, c1b, g1, be1, m1, v1, c, false);
  conv_block(hs, c2w, c2b, g2, be2, m2, v2, c, true);
  conv_block(hs, c3w, c3b, g3, be3, m3, v3, c, true);
  conv_block(hs, c4w, c4b, g4, be4, m4, v4, c, true);
  float mx = -3.4e38f;
#pragma unroll
  for (int t = 1; t <= 32; ++t) mx = fmaxf(mx, hs[c][t]);
  pooled[c] = mx;
  __syncthreads();
  if (c < 128) {
    float a = h1b[c];
    for (int i = 0; i < 256; ++i) a += pooled[i] * h1w[i * 128 + c];
    hid[c] = gelu_f(a);
  }
  __syncthreads();
  if (c < 25) {
    float a = h2b[c];
    for (int j = 0; j < 128; ++j) a += hid[j] * h2w[j * 25 + c];
    out[b * 25 + c] = a;
  }
}

// ---------------------------------------------------------------------------
extern "C" void kernel_launch(void* const* d_in, const int* in_sizes, int n_in,
                              void* d_out, int out_size, void* d_ws, size_t ws_size,
                              hipStream_t stream) {
  const float* x_pt = (const float*)d_in[0];
  const float* x_fr = (const float*)d_in[1];
  const float* ew1  = (const float*)d_in[2];
  const float* eg1  = (const float*)d_in[3];
  const float* eb1  = (const float*)d_in[4];
  const float* em1  = (const float*)d_in[5];
  const float* ev1  = (const float*)d_in[6];
  const float* ew2  = (const float*)d_in[7];
  const float* eg2  = (const float*)d_in[8];
  const float* eb2  = (const float*)d_in[9];
  const float* em2  = (const float*)d_in[10];
  const float* ev2  = (const float*)d_in[11];
  const float* p1w  = (const float*)d_in[12];
  const float* p1b  = (const float*)d_in[13];
  const float* ln1g = (const float*)d_in[14];
  const float* ln1b = (const float*)d_in[15];
  const float* p2w  = (const float*)d_in[16];
  const float* p2b  = (const float*)d_in[17];
  const float* ln2g = (const float*)d_in[18];
  const float* ln2b = (const float*)d_in[19];
  const float* f1w  = (const float*)d_in[20];
  const float* f1b  = (const float*)d_in[21];
  const float* f2w  = (const float*)d_in[22];
  const float* f2b  = (const float*)d_in[23];
  const float* projw = (const float*)d_in[24];
  const float* projb = (const float*)d_in[25];
  const float* c1w = (const float*)d_in[26];
  const float* c1b = (const float*)d_in[27];
  const float* g1  = (const float*)d_in[28];
  const float* be1 = (const float*)d_in[29];
  const float* m1  = (const float*)d_in[30];
  const float* v1  = (const float*)d_in[31];
  const float* c2w = (const float*)d_in[32];
  const float* c2b = (const float*)d_in[33];
  const float* g2  = (const float*)d_in[34];
  const float* be2 = (const float*)d_in[35];
  const float* m2  = (const float*)d_in[36];
  const float* v2  = (const float*)d_in[37];
  const float* c3w = (const float*)d_in[38];
  const float* c3b = (const float*)d_in[39];
  const float* g3  = (const float*)d_in[40];
  const float* be3 = (const float*)d_in[41];
  const float* m3  = (const float*)d_in[42];
  const float* v3  = (const float*)d_in[43];
  const float* c4w = (const float*)d_in[44];
  const float* c4b = (const float*)d_in[45];
  const float* g4  = (const float*)d_in[46];
  const float* be4 = (const float*)d_in[47];
  const float* m4  = (const float*)d_in[48];
  const float* v4  = (const float*)d_in[49];
  const float* h1w = (const float*)d_in[50];
  const float* h1b = (const float*)d_in[51];
  const float* h2w = (const float*)d_in[52];
  const float* h2b = (const float*)d_in[53];

  char* ws = (char*)d_ws;
  int*       idx     = (int*)ws;                                  //  4 MB
  _Float16*  local16 = (_Float16*)(ws + ((size_t)4  << 20));      // 16 MB
  float*     h       = (float*)   (ws + ((size_t)20 << 20));      // 64 MB
  float*     per     = (float*)   (ws + ((size_t)84 << 20));      // 0.6 MB
  const size_t fo = (size_t)85 << 20;
  _Float16* ew1f = (_Float16*)(ws + fo);                 //  4 KB (KC=1, N=64)
  _Float16* ew2f = (_Float16*)(ws + fo + (8 << 10));     // 16 KB (KC=2, N=128)
  _Float16* p1wf = (_Float16*)(ws + fo + (32 << 10));    // 40 KB (KC=5, N=128)
  _Float16* p2wf = (_Float16*)(ws + fo + (96 << 10));    // 64 KB (KC=4, N=256)
  float* out = (float*)d_out;

  // Pack weights into WMMA B-fragment layout (f16, zero padded).
  pack_wfrag_kernel<<<8,   256, 0, stream>>>(ew1, ew1f, 20, 64, 1, 1);
  pack_wfrag_kernel<<<32,  256, 0, stream>>>(ew2, ew2f, 64, 128, 2, 1);
  pack_wfrag_kernel<<<80,  256, 0, stream>>>(p1w, p1wf, 138, 128, 5, 0);
  pack_wfrag_kernel<<<128, 256, 0, stream>>>(p2w, p2wf, 128, 256, 4, 0);

  knn_kernel<<<256, 256, 0, stream>>>(x_pt, idx);
  edgeconv_kernel<<<256, 256, 0, stream>>>(x_pt, idx, ew1f, ew2f,
      eg1, eb1, em1, ev1, eg2, eb2, em2, ev2, local16);
  pointmlp_kernel<<<2048, 64, 0, stream>>>(x_pt, local16,
      p1wf, p1b, ln1g, ln1b, p2wf, p2b, ln2g, ln2b, h);
  pool_fr_kernel<<<256, 256, 0, stream>>>(h, x_fr, f1w, f1b, f2w, f2b, per);
  conv_head_kernel<<<8, 256, 0, stream>>>(per, projw, projb,
      c1w, c1b, g1, be1, m1, v1,
      c2w, c2b, g2, be2, m2, v2,
      c3w, c3b, g3, be3, m3, v3,
      c4w, c4b, g4, be4, m4, v4,
      h1w, h1b, h2w, h2b, out);
}
